// GruDecoder_19344532702330
// MI455X (gfx1250) — matrix operand order repaired and checked by reading, not compiled, not run on hardware
//
#include <hip/hip_runtime.h>
#include <math.h>

// ---------------------------------------------------------------------------
// GRU decoder for MI455X (gfx1250, wave32).
//   per step: gi = x@W_ih^T+b_ih ; gh = h@W_hh^T+b_hh ; gates ; h_new ;
//             out = [h_new|context]@fc_W^T + fc_b
//   context is constant -> precompute context@fc_W[:,H:]^T + fc_b once.
//   Matmuls: v_wmma_f32_16x16x32_bf16, f32 accumulate.
//   LDS staging: Tensor Data Mover (tensor_load_to_lds) double-buffered,
//   one barrier per K-chunk; reg-prefetch fallback if TDM builtin is absent.
// ---------------------------------------------------------------------------

typedef __bf16 bf16_t;
typedef __attribute__((ext_vector_type(16))) __bf16 v16bf;
typedef __attribute__((ext_vector_type(8)))  __bf16 v8bf;
typedef __attribute__((ext_vector_type(8)))  float  v8f;
typedef __attribute__((ext_vector_type(4)))  unsigned int u32x4;
typedef __attribute__((ext_vector_type(8)))  int i32x8;
typedef __attribute__((ext_vector_type(4)))  int i32x4;

#define GRU_B   512
#define GRU_H   1024
#define GRU_O   768
#define GRU_T   256

#define TM 64
#define TN 64
#define TK 32
#define LPAD 8   // rows padded to 40 halves (80B): conflict-free ds_load_b128

#if defined(__has_builtin)
#if __has_builtin(__builtin_amdgcn_tensor_load_to_lds) && \
    __has_builtin(__builtin_amdgcn_s_wait_tensorcnt)
#define USE_TDM 1
#endif
#endif

#ifdef USE_TDM
// ---------------------------------------------------------------------------
// Issue one TDM 2D tile load: rows x TK bf16 tile from a [rowsTot x K] row-
// major bf16 tensor into LDS, descriptor-padded +16B per 64B row (matches the
// LPAD=8 LDS layout). D# packing per CDNA5 ISA.
// This toolchain exposes the 6-arg builtin:
//   (u32x4 group0, i32x8 group1, i32x4 group2, i32x4 group3, i32x8, i32 cpol)
// ---------------------------------------------------------------------------
__device__ __forceinline__ void tdm_tile_load(const bf16_t* gsrc, void* ldsDst,
                                              int rows, int rowsTot, int K)
{
    const unsigned long long ga = (unsigned long long)(uintptr_t)gsrc;
    const unsigned lds = (unsigned)(uintptr_t)ldsDst;  // flat low bits = LDS offset
    u32x4 g0;
    g0[0] = 1u;                                           // count=1, user descriptor
    g0[1] = lds;                                          // lds_addr
    g0[2] = (unsigned)ga;                                 // global_addr[31:0]
    g0[3] = (unsigned)((ga >> 32) & 0x01FFFFFFu) | (2u << 30);  // addr[56:32] | type=2
    const unsigned td0 = (unsigned)K;        // tensor dim0 (elements per row)
    const unsigned td1 = (unsigned)rowsTot;  // tensor dim1 (rows)
    const unsigned st0 = (unsigned)K;        // dim0 stride (elements)
    i32x8 g1;
    g1[0] = (1 << 16)                         // data_size = 2 bytes
          | (1 << 20) | (3 << 22) | (3 << 25);// pad_enable, every 16 DW pad 4 DW
    g1[1] = (int)((td0 & 0xFFFFu) << 16);     // barrier_addr=0 | tensor_dim0 lo
    g1[2] = (int)((td0 >> 16) | ((td1 & 0xFFFFu) << 16));
    g1[3] = (int)((td1 >> 16) | ((unsigned)TK << 16));   // tile_dim0 = TK
    g1[4] = (int)(rows & 0xFFFF);             // tile_dim1 = rows, tile_dim2 = 0
    g1[5] = (int)st0;                         // tensor_dim0_stride[31:0]
    g1[6] = 0;                                // stride0 hi | stride1 lo (2D: unused)
    g1[7] = 0;
    const i32x4 gz4 = {0, 0, 0, 0};           // groups 2/3 unused for 2D tiles
    const i32x8 gz8 = {0, 0, 0, 0, 0, 0, 0, 0};
    __builtin_amdgcn_tensor_load_to_lds(g0, g1, gz4, gz4, gz8, 0);
}
#endif

// ---------------------------------------------------------------------------
// C[512 x N] (f32) = A[512 x K](bf16) * W[N x K](bf16)^T + bias[N] + Cadd
// Optional bf16 shadow of C. Grid (N/64, 512/64), 128 threads = 4 waves,
// each wave owns a 32x32 C tile (2x2 WMMA tiles): 8 ds_load_b128 / 4 WMMAs.
// ---------------------------------------------------------------------------
__launch_bounds__(128)
__global__ void wmma_gemm_bf16(const bf16_t* __restrict__ A,
                               const bf16_t* __restrict__ W,
                               float*        __restrict__ C,
                               bf16_t*       __restrict__ Cbf,
                               const float*  __restrict__ bias,
                               const float*  __restrict__ Cadd,
                               int K, int N)
{
    __shared__ __align__(16) bf16_t lA[2][TM][TK + LPAD];
    __shared__ __align__(16) bf16_t lW[2][TN][TK + LPAD];

    const int tid  = threadIdx.x;
    const int wave = tid >> 5;
    const int lane = tid & 31;
    const int mw   = (wave & 1) * 32;    // wave's 32-row slice
    const int nw   = (wave >> 1) * 32;   // wave's 32-col slice
    const int lr   = lane & 15;
    const int g    = lane >> 4;          // lane half -> K sub-block (ISA layout)

    const int mBase = blockIdx.y * TM;
    const int nBase = blockIdx.x * TN;

    v8f acc[2][2] = {{v8f{}, v8f{}}, {v8f{}, v8f{}}};
    int buf = 0;

#ifdef USE_TDM
    const bool tdmA = (wave == 0);       // wave0 DMAs the A tile
    const bool tdmW = (wave == 1);       // wave1 DMAs the W tile
    if (tdmA) tdm_tile_load(&A[(size_t)mBase * K], &lA[0][0][0], TM, GRU_B, K);
    if (tdmW) tdm_tile_load(&W[(size_t)nBase * K], &lW[0][0][0], TN, N, K);
#else
    // Per-thread staging: 2 chunks of 8 bf16 for A and for W.
    const int cA0 = tid, cA1 = tid + 128;
    const int rA0 = cA0 >> 2, kA0 = (cA0 & 3) * 8;
    const int rA1 = cA1 >> 2, kA1 = (cA1 & 3) * 8;
    const bf16_t* gA0 = A + (size_t)(mBase + rA0) * K + kA0;
    const bf16_t* gA1 = A + (size_t)(mBase + rA1) * K + kA1;
    const bf16_t* gW0 = W + (size_t)(nBase + rA0) * K + kA0;
    const bf16_t* gW1 = W + (size_t)(nBase + rA1) * K + kA1;
    v8bf pA0 = *(const v8bf*)gA0;
    v8bf pA1 = *(const v8bf*)gA1;
    v8bf pW0 = *(const v8bf*)gW0;
    v8bf pW1 = *(const v8bf*)gW1;
#endif

    for (int k0 = 0; k0 < K; k0 += TK) {
        const bool more = (k0 + TK) < K;
#ifdef USE_TDM
        if (wave < 2) __builtin_amdgcn_s_wait_tensorcnt((short)0);
        __syncthreads();
        if (more) {
            if (tdmA) tdm_tile_load(&A[(size_t)mBase * K + k0 + TK],
                                    &lA[buf ^ 1][0][0], TM, GRU_B, K);
            if (tdmW) tdm_tile_load(&W[(size_t)nBase * K + k0 + TK],
                                    &lW[buf ^ 1][0][0], TN, N, K);
        }
#else
        *(v8bf*)&lA[buf][rA0][kA0] = pA0;
        *(v8bf*)&lA[buf][rA1][kA1] = pA1;
        *(v8bf*)&lW[buf][rA0][kA0] = pW0;
        *(v8bf*)&lW[buf][rA1][kA1] = pW1;
        __syncthreads();
        if (more) {   // issue next-chunk global loads; consumed next iteration
            gA0 += TK; gA1 += TK; gW0 += TK; gW1 += TK;
            pA0 = *(const v8bf*)gA0;
            pA1 = *(const v8bf*)gA1;
            pW0 = *(const v8bf*)gW0;
            pW1 = *(const v8bf*)gW1;
        }
#endif
        // ---- compute from buf ----
        // A fragments (16x32): lane half g=0 holds K 0..7/16..23, g=1 8..15/24..31
        v16bf af[2];
#pragma unroll
        for (int mi = 0; mi < 2; ++mi) {
            const bf16_t* ar = &lA[buf][mw + mi * 16 + lr][0];
            v8bf alo = *(const v8bf*)(ar + 8 * g);
            v8bf ahi = *(const v8bf*)(ar + 16 + 8 * g);
#pragma unroll
            for (int i = 0; i < 8; ++i) { af[mi][i] = alo[i]; af[mi][8 + i] = ahi[i]; }
        }
        // B fragments (32x16): lane = column of W; lane half g holds K 8g..8g+7/16+8g..16+8g+7 (same as A)
#pragma unroll
        for (int ni = 0; ni < 2; ++ni) {
            const bf16_t* br = &lW[buf][nw + ni * 16 + lr][8 * g];
            v8bf blo = *(const v8bf*)(br);
            v8bf bhi = *(const v8bf*)(br + 16);
            v16bf bfr;
#pragma unroll
            for (int i = 0; i < 8; ++i) { bfr[i] = blo[i]; bfr[8 + i] = bhi[i]; }
#pragma unroll
            for (int mi = 0; mi < 2; ++mi)
                acc[mi][ni] = __builtin_amdgcn_wmma_f32_16x16x32_bf16(
                    false, af[mi], false, bfr, (short)0, acc[mi][ni], false, false);
        }
        buf ^= 1;
#ifndef USE_TDM
        __syncthreads();   // all waves must finish reading before next staging store
#endif
    }

    // Epilogue. C/D layout: element r of lane (lr + 16g) = C[m = r + 8g, n = lr].
#pragma unroll
    for (int ni = 0; ni < 2; ++ni) {
        const int n = nBase + nw + ni * 16 + lr;
        const float bv = bias ? bias[n] : 0.0f;
#pragma unroll
        for (int mi = 0; mi < 2; ++mi) {
#pragma unroll
            for (int r = 0; r < 8; ++r) {
                const int m = mBase + mw + mi * 16 + r + 8 * g;
                float v = acc[mi][ni][r] + bv;
                if (Cadd) v += Cadd[(size_t)m * N + n];
                C[(size_t)m * N + n] = v;
                if (Cbf) Cbf[(size_t)m * N + n] = (bf16_t)v;
            }
        }
    }
}

// ---------------------------------------------------------------------------
// Fused GRU gates: r,z,n + state update. 512x1024 elements.
// ---------------------------------------------------------------------------
__launch_bounds__(256)
__global__ void gru_gates(const float* __restrict__ gi, const float* __restrict__ gh,
                          const float* __restrict__ hin,
                          float* __restrict__ hout, bf16_t* __restrict__ hbf)
{
    const int idx = blockIdx.x * blockDim.x + threadIdx.x;   // 512*1024
    const int b = idx >> 10;
    const int j = idx & 1023;
    const float* gib = gi + (size_t)b * (3 * GRU_H);
    const float* ghb = gh + (size_t)b * (3 * GRU_H);
    const float ir = gib[j], iz = gib[GRU_H + j], in_ = gib[2 * GRU_H + j];
    const float hr = ghb[j], hz = ghb[GRU_H + j], hn_ = ghb[2 * GRU_H + j];
    const float r = 1.0f / (1.0f + __expf(-(ir + hr)));
    const float z = 1.0f / (1.0f + __expf(-(iz + hz)));
    const float n = tanhf(in_ + r * hn_);
    const float h = hin[idx];
    const float o = (1.0f - z) * n + z * h;
    hout[idx] = o;
    hbf[idx]  = (bf16_t)o;
}

// ---------------------------------------------------------------------------
// Strided f32 -> bf16 repack (weights once per launch; initial x/h shadows).
// ---------------------------------------------------------------------------
__launch_bounds__(256)
__global__ void f32_to_bf16(const float* __restrict__ src, bf16_t* __restrict__ dst,
                            int cols, int srcStride, int srcOff, int total)
{
    const int idx = blockIdx.x * blockDim.x + threadIdx.x;
    if (idx >= total) return;
    const int r = idx / cols;
    const int c = idx - r * cols;
    dst[idx] = (bf16_t)src[(size_t)r * srcStride + srcOff + c];
}

extern "C" void kernel_launch(void* const* d_in, const int* in_sizes, int n_in,
                              void* d_out, int out_size, void* d_ws, size_t ws_size,
                              hipStream_t stream)
{
    const float* src    = (const float*)d_in[0];   // [1,512,768]
    const float* hidden = (const float*)d_in[1];   // [1,512,1024]  (== context == h0)
    const float* W_ih   = (const float*)d_in[2];   // [3072,768]
    const float* W_hh   = (const float*)d_in[3];   // [3072,1024]
    const float* b_ih   = (const float*)d_in[4];   // [3072]
    const float* b_hh   = (const float*)d_in[5];   // [3072]
    const float* fc_W   = (const float*)d_in[6];   // [768,2048]
    const float* fc_b   = (const float*)d_in[7];   // [768]
    float*       out    = (float*)d_out;           // [256,512,768]

    const int B = GRU_B, H = GRU_H, O = GRU_O, T = GRU_T;

    // ---- workspace carve-up (~33 MB) ----
    char* ws = (char*)d_ws;
    auto carve = [&](size_t bytes) {
        char* p = ws;
        ws += (bytes + 255) & ~(size_t)255;
        return p;
    };
    bf16_t* Wih_bf  = (bf16_t*)carve((size_t)3 * H * O * 2);   // [3072 x 768]
    bf16_t* Whh_bf  = (bf16_t*)carve((size_t)3 * H * H * 2);   // [3072 x 1024]
    bf16_t* fcWh_bf = (bf16_t*)carve((size_t)O * H * 2);       // fc_W[:, :H]
    bf16_t* fcWc_bf = (bf16_t*)carve((size_t)O * H * 2);       // fc_W[:, H:]
    float*  constfc = (float*) carve((size_t)B * O * 4);       // context term + fc_b
    float*  gi      = (float*) carve((size_t)B * 3 * H * 4);
    float*  gh      = (float*) carve((size_t)B * 3 * H * 4);
    float*  h_f     = (float*) carve((size_t)B * H * 4);
    bf16_t* h_bf    = (bf16_t*)carve((size_t)B * H * 2);
    bf16_t* x_bf    = (bf16_t*)carve((size_t)B * O * 2);
    (void)ws_size; (void)in_sizes; (void)n_in; (void)out_size;

    auto cdiv = [](int a, int b) { return (a + b - 1) / b; };

    // ---- one-time (per launch) weight conversion + constant fc term ----
    f32_to_bf16<<<cdiv(3 * H * O, 256), 256, 0, stream>>>(W_ih, Wih_bf, O, O, 0, 3 * H * O);
    f32_to_bf16<<<cdiv(3 * H * H, 256), 256, 0, stream>>>(W_hh, Whh_bf, H, H, 0, 3 * H * H);
    f32_to_bf16<<<cdiv(O * H, 256), 256, 0, stream>>>(fc_W, fcWh_bf, H, 2 * H, 0, O * H);
    f32_to_bf16<<<cdiv(O * H, 256), 256, 0, stream>>>(fc_W, fcWc_bf, H, 2 * H, H, O * H);
    f32_to_bf16<<<cdiv(B * O, 256), 256, 0, stream>>>(src,    x_bf, O, O, 0, B * O);
    f32_to_bf16<<<cdiv(B * H, 256), 256, 0, stream>>>(hidden, h_bf, H, H, 0, B * H);

    // constfc = context @ fc_W[:,H:]^T + fc_b   (context == hidden[0])
    wmma_gemm_bf16<<<dim3(O / TN, B / TM), 128, 0, stream>>>(
        h_bf, fcWc_bf, constfc, nullptr, fc_b, nullptr, H, O);

    // ---- 256 sequential decoder steps ----
    for (int t = 0; t < T; ++t) {
        // gi = x @ W_ih^T + b_ih        [512 x 3072], K=768
        wmma_gemm_bf16<<<dim3(3 * H / TN, B / TM), 128, 0, stream>>>(
            x_bf, Wih_bf, gi, nullptr, b_ih, nullptr, O, 3 * H);
        // gh = h @ W_hh^T + b_hh        [512 x 3072], K=1024
        wmma_gemm_bf16<<<dim3(3 * H / TN, B / TM), 128, 0, stream>>>(
            h_bf, Whh_bf, gh, nullptr, b_hh, nullptr, H, 3 * H);
        // gates -> h_new (f32 master + bf16 shadow)
        gru_gates<<<B * H / 256, 256, 0, stream>>>(
            gi, gh, (t == 0) ? hidden : h_f, h_f, h_bf);
        // out[t] = h_new @ fc_W[:, :H]^T + constfc ; bf16 shadow becomes next x
        wmma_gemm_bf16<<<dim3(O / TN, B / TM), 128, 0, stream>>>(
            h_bf, fcWh_bf, out + (size_t)t * B * O, x_bf, nullptr, constfc, H, O);
    }
}
